// myLayer_3109556322838
// MI455X (gfx1250) — matrix operand order repaired and checked by reading, not compiled, not run on hardware
//
#include <hip/hip_runtime.h>
#include <hip/hip_bf16.h>

typedef __attribute__((ext_vector_type(16))) _Float16 v16h;
typedef __attribute__((ext_vector_type(8)))  _Float16 v8h;
typedef __attribute__((ext_vector_type(8)))  float    v8f;

#define NN   8192
#define CCH  256
#define BM   64      // block rows
#define KT   32      // K tile = one WMMA k-step

// ---------- prep kernels ----------
__global__ void gcn_prep_d(const float* __restrict__ D, float* __restrict__ dvec) {
    int i = blockIdx.x * blockDim.x + threadIdx.x;
    if (i < NN) dvec[i] = rsqrtf(D[(size_t)i * (NN + 1)]);
}

// Ht[c][k] = fp16(d[k] * H[k][c]) via LDS tile transpose (coalesced both sides)
__global__ __launch_bounds__(256) void gcn_prep_Ht(const float* __restrict__ H,
                                                   const float* __restrict__ dvec,
                                                   _Float16* __restrict__ Ht) {
    __shared__ _Float16 tile[32][33];
    const int kb = blockIdx.x * 32;          // k base (row of H)
    const int cb = blockIdx.y * 32;          // c base (col of H)
    const int tx = threadIdx.x & 31;
    const int ty = threadIdx.x >> 5;         // 0..7
    #pragma unroll
    for (int i = 0; i < 32; i += 8) {
        int k = kb + ty + i;
        tile[ty + i][tx] = (_Float16)(dvec[k] * H[(size_t)k * CCH + cb + tx]);
    }
    __syncthreads();
    #pragma unroll
    for (int i = 0; i < 32; i += 8) {
        int c = cb + ty + i;
        Ht[(size_t)c * NN + kb + tx] = tile[tx][ty + i];
    }
}

// Wt[o][i] = fp16(W[i][o])   (B matrix for GEMM2, stored [col][K]; tiny)
__global__ void gcn_prep_Wt(const float* __restrict__ W, _Float16* __restrict__ Wt) {
    int idx = blockIdx.x * blockDim.x + threadIdx.x;   // 0 .. CCH*CCH-1
    int i = idx >> 8;
    int o = idx & 255;
    Wt[(size_t)o * CCH + i] = (_Float16)W[(size_t)i * CCH + o];
}

// ---------- WMMA GEMM: C[M][256] = A[M][K] * Bt^T, Bt is [256][K] fp16 ----------
// 256 threads = 8 waves in a 2(row) x 4(col) grid; wave tile 32x64 -> acc[2][4].
// Low VGPR pressure => multiple waves/SIMD so LDS waits are hidden by co-issue.
// A_F32: A is fp32, converted to fp16 while staging (register-pipelined HBM stream).
// OUT_HALF: rows scaled by rowscale[], emitted fp16; else fp32.
template <bool A_F32, bool OUT_HALF>
__global__ __launch_bounds__(256) void gcn_gemm(
    const void* __restrict__ Ap,
    const _Float16* __restrict__ Bt,
    const float* __restrict__ rowscale,
    void* __restrict__ Cp,
    int M, int K)
{
    __shared__ __align__(16) _Float16 lA[BM * KT];     // [m][k]  4 KB
    __shared__ __align__(16) _Float16 lB[CCH * KT];    // [c][k] 16 KB

    const int tid  = threadIdx.x;
    const int wave = tid >> 5;
    const int lane = tid & 31;
    const int wr   = wave >> 2;          // wave row (0..1) -> 32 rows each
    const int wc   = wave & 3;           // wave col (0..3) -> 64 cols each
    const int row0 = blockIdx.x * BM;
    const int lrow = lane & 15;

    v8f acc[2][4] = {};                  // 32 rows x 64 cols per wave

    // ---- A-tile register staging (the HBM-resident stream) ----
    float4 aReg[2];                      // A_F32 path : 64x32 f32 = 512 float4, 2/thread
    v8h    aRegH;                        //  fp16 path : 64x32 h   = 256 16B chunks, 1/thread

    auto loadA = [&](int k0) {
        if (A_F32) {
            const float* A = (const float*)Ap;
            #pragma unroll
            for (int it = 0; it < 2; ++it) {
                int ci = tid + it * 256;
                int r  = ci >> 3;
                int c4 = ci & 7;
                aReg[it] = *(const float4*)(A + (size_t)(row0 + r) * K + k0 + c4 * 4);
            }
        } else {
            const _Float16* A = (const _Float16*)Ap;
            int r  = tid >> 2;
            int c8 = tid & 3;
            aRegH = *(const v8h*)(A + (size_t)(row0 + r) * K + k0 + c8 * 8);
        }
    };
    auto storeA = [&]() {
        if (A_F32) {
            #pragma unroll
            for (int it = 0; it < 2; ++it) {
                int ci = tid + it * 256;
                int r  = ci >> 3;
                int c4 = ci & 7;
                _Float16* dst = &lA[r * KT + c4 * 4];
                dst[0] = (_Float16)aReg[it].x; dst[1] = (_Float16)aReg[it].y;
                dst[2] = (_Float16)aReg[it].z; dst[3] = (_Float16)aReg[it].w;
            }
        } else {
            int r  = tid >> 2;
            int c8 = tid & 3;
            *(v8h*)&lA[r * KT + c8 * 8] = aRegH;
        }
    };

    union bfu { v16h v; v8h h[2]; };
    const int ka = (lane < 16) ? 0 : 16;   // A frag K sub-chunk base
    const int kb = (lane < 16) ? 0 : 16;   // B frag K base

    loadA(0);
    for (int k0 = 0; k0 < K; k0 += KT) {
        // ---- commit staged A tile; stage B tile straight from L2 ----
        storeA();
        #pragma unroll
        for (int it = 0; it < 4; ++it) {    // 1024 16B chunks, 4/thread
            int ci = tid + it * 256;
            int c  = ci >> 2;
            int j  = ci & 3;
            *(v8h*)&lB[c * KT + j * 8] =
                *(const v8h*)(Bt + (size_t)c * K + k0 + j * 8);
        }
        __syncthreads();

        // ---- kick off next A tile (HBM) so it overlaps the WMMA phase ----
        if (k0 + KT < K) loadA(k0 + KT);

        // ---- A fragments: rows wr*32 + {0,16}; contiguous K order (same as B)
        // lane<16: K 0..15; lane>=16: K 16..31
        bfu af[2];
        #pragma unroll
        for (int ai = 0; ai < 2; ++ai) {
            const _Float16* abase = &lA[(wr * 32 + ai * 16 + lrow) * KT + ka];
            af[ai].h[0] = *(const v8h*)(abase);
            af[ai].h[1] = *(const v8h*)(abase + 8);
        }

        // ---- 4 column tiles; B 32x16 layout: lane<16 K 0..15, lane>=16 K 16..31
        bfu bf[2];
        {
            const _Float16* bbase = &lB[(wc * 64 + lrow) * KT + kb];
            bf[0].h[0] = *(const v8h*)(bbase);
            bf[0].h[1] = *(const v8h*)(bbase + 8);
        }
        #pragma unroll
        for (int bi = 0; bi < 4; ++bi) {
            if (bi + 1 < 4) {
                const _Float16* bbase = &lB[(wc * 64 + (bi + 1) * 16 + lrow) * KT + kb];
                bf[(bi + 1) & 1].h[0] = *(const v8h*)(bbase);
                bf[(bi + 1) & 1].h[1] = *(const v8h*)(bbase + 8);
            }
            acc[0][bi] = __builtin_amdgcn_wmma_f32_16x16x32_f16(
                false, af[0].v, false, bf[bi & 1].v, (short)0, acc[0][bi], false, false);
            acc[1][bi] = __builtin_amdgcn_wmma_f32_16x16x32_f16(
                false, af[1].v, false, bf[bi & 1].v, (short)0, acc[1][bi], false, false);
        }
        __syncthreads();
    }

    // ---- epilogue: C 16x16 f32 layout: VGPR r -> M = r (lanes 0-15) / 8+r (16-31)
    #pragma unroll
    for (int ai = 0; ai < 2; ++ai) {
        const int mbase = row0 + wr * 32 + ai * 16 + ((lane < 16) ? 0 : 8);
        #pragma unroll
        for (int bi = 0; bi < 4; ++bi) {
            int col = wc * 64 + bi * 16 + lrow;
            #pragma unroll
            for (int r = 0; r < 8; ++r) {
                int grow = mbase + r;
                float v = acc[ai][bi][r];
                if (OUT_HALF) {
                    ((_Float16*)Cp)[(size_t)grow * CCH + col] = (_Float16)(v * rowscale[grow]);
                } else {
                    ((float*)Cp)[(size_t)grow * CCH + col] = v;
                }
            }
        }
    }
}

// ---------- launcher ----------
extern "C" void kernel_launch(void* const* d_in, const int* in_sizes, int n_in,
                              void* d_out, int out_size, void* d_ws, size_t ws_size,
                              hipStream_t stream) {
    const float* A = (const float*)d_in[0];
    const float* D = (const float*)d_in[1];
    const float* H = (const float*)d_in[2];
    const float* W = (const float*)d_in[3];
    float* out = (float*)d_out;

    char* ws = (char*)d_ws;
    float*    dvec = (float*)ws;                                   //  32 KB
    _Float16* Ht   = (_Float16*)(ws + (32u << 10));                //   4 MB : [256][8192]
    _Float16* Wt   = (_Float16*)(ws + (32u << 10) + (4u << 20));   // 128 KB : [256][256]
    _Float16* T1   = (_Float16*)(ws + (32u << 10) + (4u << 20) + (128u << 10)); // 4 MB : [8192][256]

    gcn_prep_d <<<NN / 256, 256, 0, stream>>>(D, dvec);
    gcn_prep_Ht<<<dim3(NN / 32, CCH / 32), 256, 0, stream>>>(H, dvec, Ht);
    gcn_prep_Wt<<<(CCH * CCH) / 256, 256, 0, stream>>>(W, Wt);

    // T1 = diag(d) * (A @ H')  in fp16
    gcn_gemm<true,  true ><<<NN / BM, 256, 0, stream>>>(A,  Ht, dvec,    T1,  NN, NN);
    // out = T1 @ W  in fp32
    gcn_gemm<false, false><<<NN / BM, 256, 0, stream>>>(T1, Wt, nullptr, out, NN, CCH);
}
